// MixtralMoE_59493886984630
// MI455X (gfx1250) — matrix operand order repaired and checked by reading, not compiled, not run on hardware
//
#include <hip/hip_runtime.h>
#include <math.h>

// ---------------------------------------------------------------------------
// Mixtral MoE (top-2 of 8 experts), HIDDEN=2048, INTER=4096, T=8192 tokens.
// bf16 WMMA GEMMs (v_wmma_f32_16x16x32_bf16), routed-token compaction,
// double-buffered LDS staging, async global->LDS for bf16 activation tiles.
// ---------------------------------------------------------------------------

#define T_TOK 8192
#define HID   2048
#define INTER 4096
#define NEXP  8

#define BM 128
#define BN 64
#define BK 32
#define LDA 40  // padded LDS row stride in bf16 elements (80B, 16B-aligned)

typedef __attribute__((ext_vector_type(16))) __bf16 v16bf;
typedef __attribute__((ext_vector_type(8)))  float  v8f;
typedef __attribute__((ext_vector_type(4)))  int    v4i;

__device__ __forceinline__ int imin(int a, int b) { return a < b ? a : b; }

// ---------------- async global->LDS (CDNA5 GLOBAL_LOAD_ASYNC_TO_LDS_B128) --
#if defined(__AMDGCN__) && __has_builtin(__builtin_amdgcn_global_load_async_to_lds_b128)
#define HAVE_ASYNC_LDS 1
#else
#define HAVE_ASYNC_LDS 0
#endif

typedef __attribute__((address_space(1))) v4i* gv4i_p;  // global v4i*
typedef __attribute__((address_space(3))) v4i* lv4i_p;  // LDS v4i*

__device__ __forceinline__ void cp16_to_lds(const void* g, void* l) {
#if HAVE_ASYNC_LDS
  __builtin_amdgcn_global_load_async_to_lds_b128(
      (gv4i_p)(uintptr_t)g, (lv4i_p)(unsigned)(uintptr_t)l, 0, 0);
#else
  *(uint4*)l = *(const uint4*)g;
#endif
}

__device__ __forceinline__ void async_drain() {
#if HAVE_ASYNC_LDS
#if __has_builtin(__builtin_amdgcn_s_wait_asynccnt)
  __builtin_amdgcn_s_wait_asynccnt(0);
#else
  asm volatile("s_wait_asynccnt 0x0" ::: "memory");
#endif
#endif
}

// ---------------- fragment loaders (from LDS, per ISA 7.12.2 layouts) ------
// A 16x32 bf16: lane l holds row (l&15); K = {(l>>4)*8 ..+7} U {16+(l>>4)*8 ..+7}
__device__ __forceinline__ v16bf frag_a(const __bf16* rowbase, int lane) {
  union { uint4 u[2]; v16bf v; } U;
  const int ko = (lane >> 4) * 8;
  U.u[0] = *(const uint4*)(rowbase + ko);
  U.u[1] = *(const uint4*)(rowbase + 16 + ko);
  return U.v;
}
// B 32x16 bf16: lane l holds column (l&15); K = {(l>>4)*8 ..+7} U {16+(l>>4)*8 ..+7}
__device__ __forceinline__ v16bf frag_b(const __bf16* colbase, int lane) {
  union { uint4 u[2]; v16bf v; } U;
  const int kh = (lane >> 4) * 8;
  U.u[0] = *(const uint4*)(colbase + kh);
  U.u[1] = *(const uint4*)(colbase + 16 + kh);
  return U.v;
}

__device__ __forceinline__ v8f wmma_bf16(v16bf a, v16bf b, v8f c) {
  return __builtin_amdgcn_wmma_f32_16x16x32_bf16(false, a, false, b, (short)0, c,
                                                 false, false);
}

__device__ __forceinline__ void cvt_store8(const float4& a, const float4& b,
                                           __bf16* d) {
  d[0] = (__bf16)a.x; d[1] = (__bf16)a.y; d[2] = (__bf16)a.z; d[3] = (__bf16)a.w;
  d[4] = (__bf16)b.x; d[5] = (__bf16)b.y; d[6] = (__bf16)b.z; d[7] = (__bf16)b.w;
}

// ---------------- kernel 0: zero output ------------------------------------
__global__ void moe_zero(float4* __restrict__ out) {
  out[(size_t)blockIdx.x * blockDim.x + threadIdx.x] = make_float4(0.f, 0.f, 0.f, 0.f);
}

// ---------------- kernel 1: router (1 wave / token) + X->bf16 ---------------
__global__ __launch_bounds__(256) void moe_router(
    const float* __restrict__ x, const float* __restrict__ gate,
    unsigned short* __restrict__ xb_raw, int* __restrict__ topk_e,
    float* __restrict__ topk_w, const int* __restrict__ dindex) {
  const int lane = threadIdx.x & 31;
  const int wv   = threadIdx.x >> 5;
  const int t    = blockIdx.x * 8 + wv;
  __bf16* xb = (__bf16*)xb_raw;
  const float* G = gate + (size_t)dindex[0] * NEXP * HID;

  float acc[NEXP];
#pragma unroll
  for (int e = 0; e < NEXP; ++e) acc[e] = 0.f;

  for (int k = lane; k < HID; k += 32) {
    const float xv = x[(size_t)t * HID + k];
    xb[(size_t)t * HID + k] = (__bf16)xv;  // stage bf16 activations once
#pragma unroll
    for (int e = 0; e < NEXP; ++e) acc[e] += xv * G[e * HID + k];
  }
#pragma unroll
  for (int e = 0; e < NEXP; ++e)
#pragma unroll
    for (int off = 16; off > 0; off >>= 1) acc[e] += __shfl_xor(acc[e], off, 32);

  if (lane == 0) {
    int i0 = 0;
#pragma unroll
    for (int e = 1; e < NEXP; ++e) if (acc[e] > acc[i0]) i0 = e;
    int i1 = (i0 == 0) ? 1 : 0;
#pragma unroll
    for (int e = 0; e < NEXP; ++e)
      if (e != i0 && acc[e] > acc[i1]) i1 = e;
    // renormalized top-2 softmax == softmax over the two top logits
    const float w0 = 1.f / (1.f + __expf(acc[i1] - acc[i0]));
    topk_e[2 * t] = i0;      topk_e[2 * t + 1] = i1;
    topk_w[2 * t] = w0;      topk_w[2 * t + 1] = 1.f - w0;
  }
}

// ---------------- kernel 2: deterministic compaction (1 wave / expert) ------
__global__ void moe_compact(const int* __restrict__ topk_e,
                            const float* __restrict__ topk_w,
                            int* __restrict__ lists, float* __restrict__ wts,
                            int* __restrict__ counts) {
  const int e = blockIdx.x;
  const int lane = threadIdx.x;
  int count = 0;
  for (int base = 0; base < T_TOK; base += 32) {
    const int t  = base + lane;
    const int e0 = topk_e[2 * t];
    const int e1 = topk_e[2 * t + 1];
    const bool sel = (e0 == e) || (e1 == e);
    const float w  = (e0 == e) ? topk_w[2 * t] : topk_w[2 * t + 1];
    const unsigned mask = (unsigned)__ballot(sel);
    const int pre = __popc(mask & ((1u << lane) - 1u));
    if (sel) {
      lists[e * T_TOK + count + pre] = t;
      wts[e * T_TOK + count + pre] = w;
    }
    count += __popc(mask);
  }
  if (lane == 0) counts[e] = count;
}

__global__ void moe_scan(const int* __restrict__ counts, int* __restrict__ offsets) {
  if (threadIdx.x == 0) {
    int s = 0;
    for (int e = 0; e < NEXP; ++e) { offsets[e] = s; s += counts[e]; }
    offsets[NEXP] = s;
  }
}

// ---------------- kernel 3: GEMM1  h = silu(Xg@w1^T) * (Xg@w3^T) ------------
__global__ __launch_bounds__(256) void moe_gemm1(
    const float* __restrict__ w1, const float* __restrict__ w3,
    const unsigned short* __restrict__ xb_raw, unsigned short* __restrict__ hb_raw,
    const int* __restrict__ lists, const int* __restrict__ counts,
    const int* __restrict__ offsets, const int* __restrict__ dindex) {
  const int e   = blockIdx.z;
  const int cnt = counts[e];
  const int m0  = blockIdx.x * BM;
  if (m0 >= cnt) return;
  const int n0  = blockIdx.y * BN;
  const int li  = dindex[0];
  const __bf16* xb = (const __bf16*)xb_raw;
  __bf16* hb = (__bf16*)hb_raw;
  const float* W1 = w1 + ((size_t)li * NEXP + e) * (size_t)INTER * HID;
  const float* W3 = w3 + ((size_t)li * NEXP + e) * (size_t)INTER * HID;

  __shared__ __align__(16) __bf16 As[2][BM * LDA];
  __shared__ __align__(16) __bf16 B1s[2][BN * LDA];
  __shared__ __align__(16) __bf16 B3s[2][BN * LDA];

  const int tid  = threadIdx.x;
  const int lane = tid & 31;
  const int wv   = tid >> 5;
  const int wm   = wv >> 1;  // 0..3 -> 32-row slice
  const int wn   = wv & 1;   // 0..1 -> 32-col slice
  const int seg  = tid & 3;  // 8-element (16B) segment within a 32-wide row

  // -------- hoisted, clamped gather pointers (loop-invariant) --------------
  // rows >= cnt are clamped to a valid row and masked in the epilogue.
  const int rowA0 = tid >> 2;        // 0..63
  const int rowA1 = 64 + rowA0;      // 64..127
  const int t0 = lists[e * T_TOK + imin(m0 + rowA0, cnt - 1)];
  const int t1 = lists[e * T_TOK + imin(m0 + rowA1, cnt - 1)];
  const __bf16* ga0 = xb + (size_t)t0 * HID + seg * 8;
  const __bf16* ga1 = xb + (size_t)t1 * HID + seg * 8;
  const int nB = tid >> 2;           // weight row handled by this thread
  const float* g1 = W1 + (size_t)(n0 + nB) * HID + seg * 8;
  const float* g3 = W3 + (size_t)(n0 + nB) * HID + seg * 8;
  const int ldsA0 = rowA0 * LDA + seg * 8;
  const int ldsA1 = rowA1 * LDA + seg * 8;
  const int ldsB  = nB * LDA + seg * 8;

  v8f acc1[4] = {}; v8f acc3[4] = {};

  // -------- prologue: stage k0 = 0 into buffer 0 ---------------------------
  cp16_to_lds(ga0, &As[0][ldsA0]);
  cp16_to_lds(ga1, &As[0][ldsA1]);
  {
    const float4 f1a = *(const float4*)(g1), f1b = *(const float4*)(g1 + 4);
    const float4 f3a = *(const float4*)(g3), f3b = *(const float4*)(g3 + 4);
    cvt_store8(f1a, f1b, &B1s[0][ldsB]);
    cvt_store8(f3a, f3b, &B3s[0][ldsB]);
  }
  async_drain();
  __syncthreads();

  int cur = 0;
  for (int k0 = 0; k0 < HID; k0 += BK) {
    const int nxt = cur ^ 1;
    const bool last = (k0 + BK >= HID);
    float4 f1a, f1b, f3a, f3b;
    if (!last) {
      // prefetch next tile: async bf16 A rows straight to LDS, f32 B to VGPRs
      cp16_to_lds(ga0 + k0 + BK, &As[nxt][ldsA0]);
      cp16_to_lds(ga1 + k0 + BK, &As[nxt][ldsA1]);
      f1a = *(const float4*)(g1 + k0 + BK); f1b = *(const float4*)(g1 + k0 + BK + 4);
      f3a = *(const float4*)(g3 + k0 + BK); f3b = *(const float4*)(g3 + k0 + BK + 4);
    }

    const int r = lane & 15;
    const v16bf a0 = frag_a(&As[cur][(wm * 32 + r) * LDA], lane);
    const v16bf a1 = frag_a(&As[cur][(wm * 32 + 16 + r) * LDA], lane);
    const v16bf p0 = frag_b(&B1s[cur][(wn * 32 + r) * LDA], lane);
    const v16bf p1 = frag_b(&B1s[cur][(wn * 32 + 16 + r) * LDA], lane);
    const v16bf q0 = frag_b(&B3s[cur][(wn * 32 + r) * LDA], lane);
    const v16bf q1 = frag_b(&B3s[cur][(wn * 32 + 16 + r) * LDA], lane);

    acc1[0] = wmma_bf16(a0, p0, acc1[0]);
    acc1[1] = wmma_bf16(a0, p1, acc1[1]);
    acc1[2] = wmma_bf16(a1, p0, acc1[2]);
    acc1[3] = wmma_bf16(a1, p1, acc1[3]);
    acc3[0] = wmma_bf16(a0, q0, acc3[0]);
    acc3[1] = wmma_bf16(a0, q1, acc3[1]);
    acc3[2] = wmma_bf16(a1, q0, acc3[2]);
    acc3[3] = wmma_bf16(a1, q1, acc3[3]);

    if (!last) {
      cvt_store8(f1a, f1b, &B1s[nxt][ldsB]);
      cvt_store8(f3a, f3b, &B3s[nxt][ldsB]);
      async_drain();
      __syncthreads();
    }
    cur = nxt;
  }

  // epilogue: silu(h1)*h3 -> bf16 h buffer (C/D: M=v+8*(lane>=16), N=lane&15)
  const int obase = offsets[e];
#pragma unroll
  for (int mi = 0; mi < 2; ++mi)
#pragma unroll
    for (int ni = 0; ni < 2; ++ni) {
      const v8f h1 = acc1[mi * 2 + ni];
      const v8f h3 = acc3[mi * 2 + ni];
      const int col = n0 + wn * 32 + ni * 16 + (lane & 15);
#pragma unroll
      for (int v = 0; v < 8; ++v) {
        const int rie = m0 + wm * 32 + mi * 16 + ((lane >> 4) * 8) + v;
        if (rie < cnt) {
          const float a = h1[v];
          const float hval = a * (1.f / (1.f + __expf(-a))) * h3[v];
          hb[(size_t)(obase + rie) * INTER + col] = (__bf16)hval;
        }
      }
    }
}

// ---------------- kernel 4: GEMM2  out += w_t * (h @ w2^T) ------------------
__global__ __launch_bounds__(256) void moe_gemm2(
    const float* __restrict__ w2, const unsigned short* __restrict__ hb_raw,
    float* __restrict__ out, const int* __restrict__ lists,
    const int* __restrict__ counts, const int* __restrict__ offsets,
    const float* __restrict__ wts, const int* __restrict__ dindex) {
  const int e   = blockIdx.z;
  const int cnt = counts[e];
  const int m0  = blockIdx.x * BM;
  if (m0 >= cnt) return;
  const int n0  = blockIdx.y * BN;  // over HID
  const int li  = dindex[0];
  const __bf16* hb = (const __bf16*)hb_raw;
  const float* W2 = w2 + ((size_t)li * NEXP + e) * (size_t)HID * INTER;
  const int obase = offsets[e];

  __shared__ __align__(16) __bf16 As[2][BM * LDA];
  __shared__ __align__(16) __bf16 Bs[2][BN * LDA];

  const int tid  = threadIdx.x;
  const int lane = tid & 31;
  const int wv   = tid >> 5;
  const int wm   = wv >> 1;
  const int wn   = wv & 1;
  const int seg  = tid & 3;

  const int rowA0 = tid >> 2;
  const int rowA1 = 64 + rowA0;
  const __bf16* ga0 =
      hb + (size_t)(obase + imin(m0 + rowA0, cnt - 1)) * INTER + seg * 8;
  const __bf16* ga1 =
      hb + (size_t)(obase + imin(m0 + rowA1, cnt - 1)) * INTER + seg * 8;
  const int nB = tid >> 2;
  const float* gw = W2 + (size_t)(n0 + nB) * INTER + seg * 8;
  const int ldsA0 = rowA0 * LDA + seg * 8;
  const int ldsA1 = rowA1 * LDA + seg * 8;
  const int ldsB  = nB * LDA + seg * 8;

  v8f acc[4] = {};

  cp16_to_lds(ga0, &As[0][ldsA0]);
  cp16_to_lds(ga1, &As[0][ldsA1]);
  {
    const float4 fa = *(const float4*)(gw), fb = *(const float4*)(gw + 4);
    cvt_store8(fa, fb, &Bs[0][ldsB]);
  }
  async_drain();
  __syncthreads();

  int cur = 0;
  for (int k0 = 0; k0 < INTER; k0 += BK) {
    const int nxt = cur ^ 1;
    const bool last = (k0 + BK >= INTER);
    float4 fa, fb;
    if (!last) {
      cp16_to_lds(ga0 + k0 + BK, &As[nxt][ldsA0]);
      cp16_to_lds(ga1 + k0 + BK, &As[nxt][ldsA1]);
      fa = *(const float4*)(gw + k0 + BK);
      fb = *(const float4*)(gw + k0 + BK + 4);
    }

    const int r = lane & 15;
    const v16bf a0 = frag_a(&As[cur][(wm * 32 + r) * LDA], lane);
    const v16bf a1 = frag_a(&As[cur][(wm * 32 + 16 + r) * LDA], lane);
    const v16bf b0 = frag_b(&Bs[cur][(wn * 32 + r) * LDA], lane);
    const v16bf b1 = frag_b(&Bs[cur][(wn * 32 + 16 + r) * LDA], lane);

    acc[0] = wmma_bf16(a0, b0, acc[0]);
    acc[1] = wmma_bf16(a0, b1, acc[1]);
    acc[2] = wmma_bf16(a1, b0, acc[2]);
    acc[3] = wmma_bf16(a1, b1, acc[3]);

    if (!last) {
      cvt_store8(fa, fb, &Bs[nxt][ldsB]);
      async_drain();
      __syncthreads();
    }
    cur = nxt;
  }

#pragma unroll
  for (int mi = 0; mi < 2; ++mi)
#pragma unroll
    for (int ni = 0; ni < 2; ++ni) {
      const v8f c = acc[mi * 2 + ni];
      const int col = n0 + wn * 32 + ni * 16 + (lane & 15);
#pragma unroll
      for (int v = 0; v < 8; ++v) {
        const int rie = m0 + wm * 32 + mi * 16 + ((lane >> 4) * 8) + v;
        if (rie < cnt) {
          const int   t  = lists[e * T_TOK + rie];
          const float sw = wts[e * T_TOK + rie];
          atomicAdd(&out[(size_t)t * HID + col], c[v] * sw);
        }
      }
    }
}

// ---------------------------------------------------------------------------
extern "C" void kernel_launch(void* const* d_in, const int* in_sizes, int n_in,
                              void* d_out, int out_size, void* d_ws, size_t ws_size,
                              hipStream_t stream) {
  const int*   d_index = (const int*)d_in[0];
  const float* x    = (const float*)d_in[1];
  const float* gate = (const float*)d_in[2];
  const float* w1   = (const float*)d_in[3];
  const float* w3   = (const float*)d_in[4];
  const float* w2   = (const float*)d_in[5];
  float* out = (float*)d_out;

  char* ws = (char*)d_ws;
  // workspace layout (~161 MB total)
  int*   counts  = (int*)(ws + 0);
  int*   offsets = (int*)(ws + 1024);
  int*   lists   = (int*)(ws + 2048);                      // 8*8192*4  = 256 KB
  float* wts     = (float*)(ws + 2048 + 262144);           // 256 KB
  int*   tke     = (int*)(ws + 2048 + 2 * 262144);         // 64 KB
  float* tkw     = (float*)(ws + 2048 + 2 * 262144 + 65536);
  unsigned short* xb = (unsigned short*)(ws + 657408);     // T*H bf16 = 32 MB
  unsigned short* hb = (unsigned short*)(ws + 657408 + (size_t)T_TOK * HID * 2);
  (void)in_sizes; (void)n_in; (void)out_size; (void)ws_size;

  moe_zero<<<(T_TOK * (size_t)HID / 4) / 256, 256, 0, stream>>>((float4*)out);
  moe_router<<<T_TOK / 8, 256, 0, stream>>>(x, gate, xb, tke, tkw, d_index);
  moe_compact<<<NEXP, 32, 0, stream>>>(tke, tkw, lists, wts, counts);
  moe_scan<<<1, 32, 0, stream>>>(counts, offsets);
  moe_gemm1<<<dim3(T_TOK / BM, INTER / BN, NEXP), 256, 0, stream>>>(
      w1, w3, xb, hb, lists, counts, offsets, d_index);
  moe_gemm2<<<dim3(T_TOK / BM, HID / BN, NEXP), 256, 0, stream>>>(
      w2, hb, out, lists, counts, offsets, wts, d_index);
}
